// AttModel_61134564491521
// MI455X (gfx1250) — matrix operand order repaired and checked by reading, not compiled, not run on hardware
//
#include <hip/hip_runtime.h>
#include <hip/hip_bf16.h>

typedef __attribute__((ext_vector_type(16))) _Float16 v16h;
typedef __attribute__((ext_vector_type(8)))  _Float16 v8h;
typedef __attribute__((ext_vector_type(8)))  float    v8f;

#define BN 16
#define NN 2048
#define DD 128
#define HH 128
static constexpr float NEG_BIAS = 9000000000000000.0f;

// ---------------------------------------------------------------------------
// Kernel 0: W[d][h] f32 -> Wt[h][d] f16 (so WMMA B-operand loads are 32B/lane)
// ---------------------------------------------------------------------------
__global__ void prep_weights(const float* __restrict__ Wq, const float* __restrict__ Wk,
                             const float* __restrict__ Wv,
                             _Float16* __restrict__ Wqt, _Float16* __restrict__ Wkt,
                             _Float16* __restrict__ Wvt) {
    int t = blockIdx.x * blockDim.x + threadIdx.x;
    if (t < DD * HH) {
        int d = t / HH, h = t % HH;
        Wqt[h * DD + d] = (_Float16)Wq[t];
        Wkt[h * DD + d] = (_Float16)Wk[t];
        Wvt[h * DD + d] = (_Float16)Wv[t];
    }
}

// ---------------------------------------------------------------------------
// Kernel 1: QKV projection with f16 WMMA, relu(x@W + b).
// proj is encoded in blockIdx.x (scalar -> uniform s_cbranch, no exec juggling).
// Q,K stored [b][n][h] f16; V stored transposed [b][h][n] f16.
// ---------------------------------------------------------------------------
__global__ __launch_bounds__(128) void qkv_proj(
    const float* __restrict__ x,
    const _Float16* __restrict__ Wqt, const _Float16* __restrict__ Wkt,
    const _Float16* __restrict__ Wvt,
    const float* __restrict__ bq, const float* __restrict__ bk, const float* __restrict__ bv,
    _Float16* __restrict__ Qh, _Float16* __restrict__ Kh, _Float16* __restrict__ Vt)
{
    const int BLKS_PER_PROJ = BN * (NN / 16) / 4;          // 512
    const int proj = blockIdx.x / BLKS_PER_PROJ;           // scalar: 0=Q 1=K 2=V
    const int lane = threadIdx.x & 31;
    const int wave = threadIdx.x >> 5;
    const int rem  = (blockIdx.x % BLKS_PER_PROJ) * 4 + wave;  // 0 .. BN*(NN/16)-1
    const int b    = rem / (NN / 16);
    const int i0   = (rem % (NN / 16)) * 16;
    const int hlf  = lane >> 4;                            // half-wave select
    const int lr   = lane & 15;

    const _Float16* Wt   = (proj == 0) ? Wqt : (proj == 1) ? Wkt : Wvt;
    const float*    bias = (proj == 0) ? bq  : (proj == 1) ? bk  : bv;
    _Float16*       dstQK = (proj == 0) ? Qh : Kh;

    // A-operand: x rows in f16.  A 16x32 layout: lanes<16 K={c*32+0..15},
    // lanes>=16 K={c*32+16..31}; row M = lane&15.
    v16h a[4];
    const float* xrow = x + ((size_t)b * NN + i0 + lr) * DD;
#pragma unroll
    for (int c = 0; c < 4; ++c) {
        const int k0 = c * 32 + hlf * 16;
#pragma unroll
        for (int e = 0; e < 8; ++e) {
            a[c][e]     = (_Float16)xrow[k0 + e];
            a[c][8 + e] = (_Float16)xrow[k0 + 8 + e];
        }
    }

#pragma unroll
    for (int ht = 0; ht < 8; ++ht) {
        const int h0 = ht * 16;
        v8f acc = {};
#pragma unroll
        for (int c = 0; c < 4; ++c) {
            // B 32x16: lane half holds 16 sequential K for its column h0+lr
            v16h bm = *(const v16h*)(Wt + (size_t)(h0 + lr) * DD + c * 32 + hlf * 16);
            acc = __builtin_amdgcn_wmma_f32_16x16x32_f16(false, a[c], false, bm,
                                                         (short)0, acc, false, false);
        }
        const float bb = bias[h0 + lr];
        _Float16 vals[8];
#pragma unroll
        for (int r = 0; r < 8; ++r) {
            float v = acc[r] + bb;
            vals[r] = (_Float16)(v > 0.0f ? v : 0.0f);     // relu
        }
        if (proj == 2) {                                   // uniform scalar branch
#pragma unroll
            for (int r = 0; r < 8; ++r) {
                const int row = i0 + r + 8 * hlf;          // D layout row
                Vt[((size_t)b * HH + h0 + lr) * NN + row] = vals[r];
            }
        } else {
#pragma unroll
            for (int r = 0; r < 8; ++r) {
                const int row = i0 + r + 8 * hlf;
                dstQK[((size_t)b * NN + row) * HH + h0 + lr] = vals[r];
            }
        }
    }
}

// ---------------------------------------------------------------------------
// Kernel 2: flash attention. One wave = one (b, 16-row i-tile); loop over j
// in chunks of 32 with online softmax. Mask streamed with non-temporal loads
// (268MB, read once) so f16 K/V (1MB per batch) stay L2-resident.
// ---------------------------------------------------------------------------
__global__ __launch_bounds__(128) void flash_attn(
    const _Float16* __restrict__ Qh, const _Float16* __restrict__ Kh,
    const _Float16* __restrict__ Vt, const float* __restrict__ mask,
    float* __restrict__ out)
{
    __shared__ alignas(16) _Float16 pbuf[4][16][40];   // per-wave P transpose tile
    const int lane = threadIdx.x & 31;
    const int wave = threadIdx.x >> 5;
    const int wid  = blockIdx.x * 4 + wave;            // 0 .. BN*(NN/16)-1
    const int b    = wid / (NN / 16);
    const int i0   = (wid % (NN / 16)) * 16;
    const int hlf  = lane >> 4;
    const int lr   = lane & 15;

    // Q A-operands for the whole 16x128 tile (4 K-chunks of 32)
    v16h qa[4];
    const _Float16* qrow = Qh + ((size_t)b * NN + i0 + lr) * HH;
#pragma unroll
    for (int c = 0; c < 4; ++c) {
        const int k0 = c * 32 + hlf * 16;
        v8h lo = *(const v8h*)(qrow + k0);
        v8h hi = *(const v8h*)(qrow + k0 + 8);
#pragma unroll
        for (int e = 0; e < 8; ++e) { qa[c][e] = lo[e]; qa[c][8 + e] = hi[e]; }
    }

    v8f zero = {};
    v8f acc[8];
#pragma unroll
    for (int t = 0; t < 8; ++t) acc[t] = zero;
    float rmax[8], rsum[8];
#pragma unroll
    for (int r = 0; r < 8; ++r) { rmax[r] = -3.0e38f; rsum[r] = 0.0f; }

    const _Float16* Kb = Kh + (size_t)b * NN * HH;
    const _Float16* Vb = Vt + (size_t)b * HH * NN;
    const float*    Mb = mask + ((size_t)b * NN + i0) * NN;

    for (int j0 = 0; j0 < NN; j0 += 32) {
        // ---- S = Q K^T for two 16-wide j tiles (K-dim 128 = 4 x 32) ----
        v8f s0 = zero, s1 = zero;
#pragma unroll
        for (int c = 0; c < 4; ++c) {
            v16h kb = *(const v16h*)(Kb + (size_t)(j0 + lr) * HH + c * 32 + hlf * 16);
            s0 = __builtin_amdgcn_wmma_f32_16x16x32_f16(false, qa[c], false, kb,
                                                        (short)0, s0, false, false);
        }
#pragma unroll
        for (int c = 0; c < 4; ++c) {
            v16h kb = *(const v16h*)(Kb + (size_t)(j0 + 16 + lr) * HH + c * 32 + hlf * 16);
            s1 = __builtin_amdgcn_wmma_f32_16x16x32_f16(false, qa[c], false, kb,
                                                        (short)0, s1, false, false);
        }

        // ---- mask + online softmax (rows live per-VGPR, cols per-lane) ----
        float p0[8], p1[8];
#pragma unroll
        for (int r = 0; r < 8; ++r) {
            const float* mrow = Mb + (size_t)(r + 8 * hlf) * NN + j0 + lr;
            const float m0 = __builtin_nontemporal_load(mrow);
            const float m1 = __builtin_nontemporal_load(mrow + 16);
            const float l0 = s0[r] * m0 - NEG_BIAS * (1.0f - m0);
            const float l1 = s1[r] * m1 - NEG_BIAS * (1.0f - m1);
            float tm = fmaxf(l0, l1);
#pragma unroll
            for (int off = 8; off > 0; off >>= 1)
                tm = fmaxf(tm, __shfl_xor(tm, off, 32));   // reduce within 16-lane half
            const float nm   = fmaxf(rmax[r], tm);
            const float corr = __expf(rmax[r] - nm);
            rmax[r] = nm;
            const float e0 = __expf(l0 - nm);
            const float e1 = __expf(l1 - nm);
            float ps = e0 + e1;
#pragma unroll
            for (int off = 8; off > 0; off >>= 1)
                ps += __shfl_xor(ps, off, 32);
            rsum[r] = rsum[r] * corr + ps;
            p0[r] = e0; p1[r] = e1;
#pragma unroll
            for (int t = 0; t < 8; ++t) acc[t][r] *= corr;  // rescale O rows
        }

        // ---- transpose P: D layout -> A layout via per-wave LDS tile ----
        // wave_barrier: WAR fence so these stores cannot be scheduled above the
        // previous iteration's cross-lane loads (hardware DS is in-order per wave;
        // these barriers only constrain compiler scheduling, emitting nothing).
        __builtin_amdgcn_wave_barrier();
#pragma unroll
        for (int r = 0; r < 8; ++r) {
            const int row = r + 8 * hlf;
            pbuf[wave][row][lr]      = (_Float16)p0[r];
            pbuf[wave][row][16 + lr] = (_Float16)p1[r];
        }
        // RAW fence: loads below read lanes' data written above.
        __builtin_amdgcn_wave_barrier();
        v16h pa;
        {
            const int k0 = hlf * 16;
#pragma unroll
            for (int e = 0; e < 8; ++e) {
                pa[e]     = pbuf[wave][lr][k0 + e];
                pa[8 + e] = pbuf[wave][lr][k0 + 8 + e];
            }
        }

        // ---- O += P * V  (V^T layout makes B-operand contiguous 32B/lane) ----
#pragma unroll
        for (int t = 0; t < 8; ++t) {
            v16h vb = *(const v16h*)(Vb + (size_t)(t * 16 + lr) * NN + j0 + hlf * 16);
            acc[t] = __builtin_amdgcn_wmma_f32_16x16x32_f16(false, pa, false, vb,
                                                            (short)0, acc[t], false, false);
        }
    }

    // ---- normalize and write out [b][n][h] f32 (write-once -> non-temporal) ----
#pragma unroll
    for (int t = 0; t < 8; ++t) {
#pragma unroll
        for (int r = 0; r < 8; ++r) {
            const int row = i0 + r + 8 * hlf;
            __builtin_nontemporal_store(acc[t][r] / rsum[r],
                out + ((size_t)b * NN + row) * HH + t * 16 + lr);
        }
    }
}

// ---------------------------------------------------------------------------
extern "C" void kernel_launch(void* const* d_in, const int* in_sizes, int n_in,
                              void* d_out, int out_size, void* d_ws, size_t ws_size,
                              hipStream_t stream) {
    (void)in_sizes; (void)n_in; (void)out_size; (void)ws_size;
    const float* x    = (const float*)d_in[0];
    const float* mask = (const float*)d_in[1];
    const float* Wv   = (const float*)d_in[2];
    const float* bv   = (const float*)d_in[3];
    const float* Wk   = (const float*)d_in[4];
    const float* bk   = (const float*)d_in[5];
    const float* Wq   = (const float*)d_in[6];
    const float* bq   = (const float*)d_in[7];
    float* out = (float*)d_out;

    _Float16* Qh  = (_Float16*)d_ws;                       // [B][N][H]
    _Float16* Kh  = Qh  + (size_t)BN * NN * HH;            // [B][N][H]
    _Float16* Vt  = Kh  + (size_t)BN * NN * HH;            // [B][H][N]
    _Float16* Wqt = Vt  + (size_t)BN * NN * HH;            // [H][D]
    _Float16* Wkt = Wqt + (size_t)DD * HH;
    _Float16* Wvt = Wkt + (size_t)DD * HH;

    prep_weights<<<(DD * HH + 255) / 256, 256, 0, stream>>>(Wq, Wk, Wv, Wqt, Wkt, Wvt);
    qkv_proj<<<3 * (BN * (NN / 16) / 4), 128, 0, stream>>>(x, Wqt, Wkt, Wvt,
                                                           bq, bk, bv, Qh, Kh, Vt);
    flash_attn<<<BN * (NN / 16) / 4, 128, 0, stream>>>(Qh, Kh, Vt, mask, out);
}
